// TritonGatherConv_55387898250089
// MI455X (gfx1250) — matrix operand order repaired and checked by reading, not compiled, not run on hardware
//
#include <hip/hip_runtime.h>

// ---------------- problem constants (fixed shapes from the reference) ------
#define H_      16
#define K_      7
#define HALF_S_ 3
#define MAXF_   8.0f
#define MINF_   0.5f
#define EPS_    1e-5f
#define B_      2
#define L_      4096
#define C_      1024
#define D_      64              // C_/H_
#define NT_     (B_*L_)         // 8192 tokens
#define NF_     144             // fused front-GEMM width: 32 (wave) + 112 (kernel)
#define KS_     64              // K elements staged to LDS per step (2 WMMA sub-steps)

typedef __bf16 bf16;
typedef __bf16 v16bf __attribute__((ext_vector_type(16)));
typedef __bf16 v8bf  __attribute__((ext_vector_type(8)));
typedef float  v8f   __attribute__((ext_vector_type(8)));
typedef unsigned int u32x4 __attribute__((ext_vector_type(4)));
typedef int          i32x4 __attribute__((ext_vector_type(4)));
typedef int          i32x8 __attribute__((ext_vector_type(8)));

union AF { v16bf v; v8bf h[2]; };

__device__ __forceinline__ float sigm_(float x) { return 1.0f / (1.0f + __expf(-x)); }
__device__ __forceinline__ float silu_(float x) { return x * sigm_(x); }

#if __has_builtin(__builtin_amdgcn_tensor_load_to_lds) && __has_builtin(__builtin_amdgcn_s_wait_tensorcnt)
#define HAVE_TDM 1
#else
#define HAVE_TDM 0
#endif

#if HAVE_TDM
// Issue one TDM 2-D tile load: global (row-major, 2-byte elems, row stride
// `stride0` elems) -> LDS (tile packed row-major, tile_d0*2 bytes per row).
// Descriptor bitfields per CDNA5 ISA ch.8 (D# group0/group1).
__device__ __forceinline__ void tdm_load_2d(const bf16* gsrc, bf16* lds,
                                            unsigned tensor_d0, unsigned tensor_d1,
                                            unsigned tile_d0, unsigned tile_d1,
                                            unsigned stride0) {
    unsigned long long ga = (unsigned long long)(size_t)gsrc;   // byte address
    unsigned lo = (unsigned)(size_t)(void*)lds;                 // LDS byte offset (low 32b of aperture addr)

    u32x4 g0;
    g0[0] = 1u;                                            // count=1, user descriptor
    g0[1] = lo;                                            // lds_addr
    g0[2] = (unsigned)(ga & 0xFFFFFFFFu);                  // global_addr[31:0]
    g0[3] = (unsigned)((ga >> 32) & 0x1FFFFFFu) | (2u << 30); // global_addr[56:32] | type=2

    i32x8 g1;
    g1[0] = (int)(1u << 16);                               // workgroup_mask=0, data_size=1 (2B)
    g1[1] = (int)((tensor_d0 & 0xFFFFu) << 16);            // tensor_dim0[15:0]
    g1[2] = (int)(((tensor_d0 >> 16) & 0xFFFFu) | ((tensor_d1 & 0xFFFFu) << 16));
    g1[3] = (int)(((tensor_d1 >> 16) & 0xFFFFu) | ((tile_d0 & 0xFFFFu) << 16));
    g1[4] = (int)(tile_d1 & 0xFFFFu);                      // tile_dim1, tile_dim2=0
    g1[5] = (int)stride0;                                  // tensor_dim0_stride[31:0] (elems)
    g1[6] = 0;                                             // stride0[47:32]=0, stride1 lo=0
    g1[7] = 0;

    i32x4 z4 = {0, 0, 0, 0};
#if defined(__clang_major__) && (__clang_major__ >= 23)
    i32x8 z8 = {0, 0, 0, 0, 0, 0, 0, 0};
    __builtin_amdgcn_tensor_load_to_lds(g0, g1, z4, z4, z8, 0);
#else
    __builtin_amdgcn_tensor_load_to_lds(g0, g1, z4, z4, 0);
#endif
}
#endif

// ---------------- 0) f32 -> bf16 conversion (grid-stride) ------------------
__global__ void cvt_bf16(const float* __restrict__ src, bf16* __restrict__ dst, int n) {
    int i = blockIdx.x * blockDim.x + threadIdx.x;
    int stride = gridDim.x * blockDim.x;
    for (; i < n; i += stride) dst[i] = (bf16)src[i];
}

// ---------------- 1) front GEMM (8192x1024 @ 1024x144) + RMSNorm + SiLU ----
// 128 threads = 4 waves; block tile 64x144; K staged 64-wide through LDS by
// the TDM, double-buffered; 18 WMMAs per wave per barrier window.
__global__ void __launch_bounds__(128)
gemm_front(const bf16* __restrict__ X,    // [NT_][C_] bf16
           const bf16* __restrict__ Wf,   // [NF_][C_] bf16 (rows 0..31 wave_w, 32..143 kernel_w)
           const float* __restrict__ wave_b, const float* __restrict__ wave_g,
           const float* __restrict__ kern_b, const float* __restrict__ kern_g,
           float* __restrict__ freq, float* __restrict__ phase, float* __restrict__ kern) {
    __shared__ float sbuf[4][16][NF_ + 16];   // epilogue transpose buffer (40KB)
    __shared__ bf16 sA[2][64 * KS_];          // 2 x 8KB   (A tile: 64 rows x 64 K)
    __shared__ bf16 sB[2][NF_ * KS_];         // 2 x 18KB  (B tile: 144 rows x 64 K)

    const int lane   = threadIdx.x & 31;
    const int wv     = threadIdx.x >> 5;
    const int tile_m = blockIdx.x * 4 + wv;   // 16-row strip per wave
    const int m_base = blockIdx.x * 64;       // block M base
    const int lr     = lane & 15;
    const int sel    = lane >> 4;

    v8f acc[9] = {};

#if HAVE_TDM
    if (wv == 0) {
        tdm_load_2d(X  + (size_t)m_base * C_, sA[0], C_, NT_, KS_, 64,  C_);
        tdm_load_2d(Wf,                       sB[0], C_, NF_, KS_, NF_, C_);
        __builtin_amdgcn_s_wait_tensorcnt(0);
    }
    __syncthreads();
#else
    {
        for (int c = threadIdx.x; c < 64 * 8; c += 128)
            *(uint4*)(sA[0] + (c >> 3) * KS_ + (c & 7) * 8) =
                *(const uint4*)(X + (size_t)(m_base + (c >> 3)) * C_ + (c & 7) * 8);
        for (int c = threadIdx.x; c < NF_ * 8; c += 128)
            *(uint4*)(sB[0] + (c >> 3) * KS_ + (c & 7) * 8) =
                *(const uint4*)(Wf + (size_t)(c >> 3) * C_ + (c & 7) * 8);
        __syncthreads();
    }
#endif

    for (int it = 0; it < C_ / KS_; ++it) {
        const int p     = it & 1;
        const int knxt  = (it + 1) * KS_;
        const bool more = knxt < C_;

#if HAVE_TDM
        if (wv == 0 && more) {
            tdm_load_2d(X  + (size_t)m_base * C_ + knxt, sA[p ^ 1], C_, NT_, KS_, 64,  C_);
            tdm_load_2d(Wf + knxt,                       sB[p ^ 1], C_, NF_, KS_, NF_, C_);
        }
#else
        if (more) {
            for (int c = threadIdx.x; c < 64 * 8; c += 128)
                *(uint4*)(sA[p ^ 1] + (c >> 3) * KS_ + (c & 7) * 8) =
                    *(const uint4*)(X + (size_t)(m_base + (c >> 3)) * C_ + knxt + (c & 7) * 8);
            for (int c = threadIdx.x; c < NF_ * 8; c += 128)
                *(uint4*)(sB[p ^ 1] + (c >> 3) * KS_ + (c & 7) * 8) =
                    *(const uint4*)(Wf + (size_t)(c >> 3) * C_ + knxt + (c & 7) * 8);
        }
#endif

#pragma unroll
        for (int ks = 0; ks < KS_; ks += 32) {
            AF a;
            const bf16* arow = sA[p] + (wv * 16 + lr) * KS_ + ks;
            a.h[0] = *(const v8bf*)(arow + 16 * sel);
            a.h[1] = *(const v8bf*)(arow + 16 * sel + 8);

            v16bf bfr[9];
#pragma unroll
            for (int t = 0; t < 9; ++t)
                bfr[t] = *(const v16bf*)(sB[p] + (t * 16 + lr) * KS_ + ks + 16 * sel);
#pragma unroll
            for (int t = 0; t < 9; ++t)
                acc[t] = __builtin_amdgcn_wmma_f32_16x16x32_bf16(
                    false, a.v, false, bfr[t], (short)0, acc[t], false, false);
        }

#if HAVE_TDM
        if (wv == 0 && more) __builtin_amdgcn_s_wait_tensorcnt(0);
#endif
        __syncthreads();
    }

    // dump C tile to LDS: VGPR i -> row i + 8*sel, col lane&15
    float (*s)[NF_ + 16] = sbuf[wv];
#pragma unroll
    for (int t = 0; t < 9; ++t)
#pragma unroll
        for (int i = 0; i < 8; ++i)
            s[i + 8 * sel][t * 16 + lr] = acc[t][i];
    __syncthreads();

    const int r     = lr;
    const int token = tile_m * 16 + r;
    if (lane < 16) {
        float y[32];
        float ss = 0.0f;
#pragma unroll
        for (int j = 0; j < 32; ++j) { float v = s[r][j] + wave_b[j]; y[j] = v; ss += v * v; }
        float rr = rsqrtf(ss * (1.0f / 32.0f) + EPS_);
#pragma unroll
        for (int h = 0; h < H_; ++h) {
            float f = y[h] * rr * wave_g[h];            f = silu_(f);
            float p = y[16 + h] * rr * wave_g[16 + h];  p = silu_(p);
            freq [token * H_ + h] = sigm_(f) * (MAXF_ - MINF_) + MINF_;
            phase[token * H_ + h] = tanhf(p) * MAXF_;
        }
    } else {
        float ss = 0.0f;
        for (int j = 0; j < 112; ++j) { float v = s[r][32 + j] + kern_b[j]; ss += v * v; }
        float rr = rsqrtf(ss * (1.0f / 112.0f) + EPS_);
        for (int j = 0; j < 112; ++j) {
            float v = (s[r][32 + j] + kern_b[j]) * rr * kern_g[j];
            kern[token * 112 + j] = silu_(v);
        }
    }
}

// ---------------- 2) fractional gather + tap contraction -------------------
__global__ void __launch_bounds__(256)
gather_conv(const float* __restrict__ x, const float* __restrict__ freq,
            const float* __restrict__ phase, const float* __restrict__ kern,
            bf16* __restrict__ hidden) {
    const int lane = threadIdx.x & 31;
    const int w    = (blockIdx.x * blockDim.x + threadIdx.x) >> 5;
    const int h    = w & (H_ - 1);
    const int tok  = w >> 4;
    const int l    = tok & (L_ - 1);

    const float fr = freq[w];
    const float ph = phase[w];
    const float* xb = x + (size_t)(tok - l) * C_;

    float a0 = 0.0f, a1 = 0.0f;
#pragma unroll
    for (int k = 0; k < K_; ++k) {
        float rel = (float)(k - HALF_S_) * fr + ph;
        rel = fminf(fmaxf(rel, -64.0f), 64.0f);
        float src = fminf(fmaxf((float)l + rel, 0.0f), (float)(L_ - 1));
        float lof = floorf(src);
        int   lo  = (int)lof;
        float fc  = src - lof;
        int   hi  = min(lo + 1, L_ - 1);
        const float* plo = xb + (size_t)lo * C_ + h * D_;
        const float* phi = xb + (size_t)hi * C_ + h * D_;
        float kv = kern[w * K_ + k];
        float v0 = plo[lane], v1 = plo[lane + 32];
        float u0 = phi[lane], u1 = phi[lane + 32];
        a0 += kv * (v0 + fc * (u0 - v0));
        a1 += kv * (v1 + fc * (u1 - v1));
    }
    bf16* hout = hidden + (size_t)tok * C_ + h * D_;
    hout[lane]      = (bf16)a0;
    hout[lane + 32] = (bf16)a1;
}

// ---------------- 3) out GEMM (8192x1024 @ 1024x1024) + SiLU ---------------
// 256 threads = 8 waves; block tile 128x128; K staged 64-wide through LDS by
// the TDM, double-buffered; 16 WMMAs per wave per barrier window.
__global__ void __launch_bounds__(256)
gemm_out(const bf16* __restrict__ Hd,   // [NT_][C_]
         const bf16* __restrict__ Wo,   // [C_][C_] (N x K row-major)
         float* __restrict__ out) {
    __shared__ bf16 sA[2][128 * KS_];   // 2 x 16KB
    __shared__ bf16 sB[2][128 * KS_];   // 2 x 16KB

    const int lane   = threadIdx.x & 31;
    const int wv     = threadIdx.x >> 5;
    const int bm     = blockIdx.x >> 3;
    const int bn     = blockIdx.x & 7;
    const int m_base = bm * 128;
    const int n_base = bn * 128;
    const int lr     = lane & 15;
    const int sel    = lane >> 4;

    v8f acc[8] = {};

#if HAVE_TDM
    if (wv == 0) {
        tdm_load_2d(Hd + (size_t)m_base * C_, sA[0], C_, NT_, KS_, 128, C_);
        tdm_load_2d(Wo + (size_t)n_base * C_, sB[0], C_, C_,  KS_, 128, C_);
        __builtin_amdgcn_s_wait_tensorcnt(0);
    }
    __syncthreads();
#else
    {   // cooperative staging fallback: 128 rows x 8 chunks of 16B per tile
        for (int c = threadIdx.x; c < 128 * 8; c += 256) {
            *(uint4*)(sA[0] + (c >> 3) * KS_ + (c & 7) * 8) =
                *(const uint4*)(Hd + (size_t)(m_base + (c >> 3)) * C_ + (c & 7) * 8);
            *(uint4*)(sB[0] + (c >> 3) * KS_ + (c & 7) * 8) =
                *(const uint4*)(Wo + (size_t)(n_base + (c >> 3)) * C_ + (c & 7) * 8);
        }
        __syncthreads();
    }
#endif

    for (int it = 0; it < C_ / KS_; ++it) {
        const int p     = it & 1;
        const int knxt  = (it + 1) * KS_;
        const bool more = knxt < C_;

#if HAVE_TDM
        if (wv == 0 && more) {
            tdm_load_2d(Hd + (size_t)m_base * C_ + knxt, sA[p ^ 1], C_, NT_, KS_, 128, C_);
            tdm_load_2d(Wo + (size_t)n_base * C_ + knxt, sB[p ^ 1], C_, C_,  KS_, 128, C_);
        }
#else
        if (more) {
            for (int c = threadIdx.x; c < 128 * 8; c += 256) {
                *(uint4*)(sA[p ^ 1] + (c >> 3) * KS_ + (c & 7) * 8) =
                    *(const uint4*)(Hd + (size_t)(m_base + (c >> 3)) * C_ + knxt + (c & 7) * 8);
                *(uint4*)(sB[p ^ 1] + (c >> 3) * KS_ + (c & 7) * 8) =
                    *(const uint4*)(Wo + (size_t)(n_base + (c >> 3)) * C_ + knxt + (c & 7) * 8);
            }
        }
#endif

#pragma unroll
        for (int ks = 0; ks < KS_; ks += 32) {
            // A fragment from LDS (contiguous k = 16*sel + s, same order as B)
            AF a;
            const bf16* arow = sA[p] + (wv * 16 + lr) * KS_ + ks;
            a.h[0] = *(const v8bf*)(arow + 16 * sel);
            a.h[1] = *(const v8bf*)(arow + 16 * sel + 8);

            // all 8 B fragments from LDS, then the WMMA chain
            v16bf bfr[8];
#pragma unroll
            for (int t = 0; t < 8; ++t)
                bfr[t] = *(const v16bf*)(sB[p] + (t * 16 + lr) * KS_ + ks + 16 * sel);
#pragma unroll
            for (int t = 0; t < 8; ++t)
                acc[t] = __builtin_amdgcn_wmma_f32_16x16x32_bf16(
                    false, a.v, false, bfr[t], (short)0, acc[t], false, false);
        }

#if HAVE_TDM
        if (wv == 0 && more) __builtin_amdgcn_s_wait_tensorcnt(0);
#endif
        __syncthreads();
    }

#pragma unroll
    for (int t = 0; t < 8; ++t)
#pragma unroll
        for (int i = 0; i < 8; ++i) {
            int m = m_base + wv * 16 + i + 8 * sel;
            int n = n_base + t * 16 + lr;
            float v = acc[t][i];
            out[(size_t)m * C_ + n] = v * sigm_(v);
        }
}

// ---------------------------------------------------------------------------
extern "C" void kernel_launch(void* const* d_in, const int* in_sizes, int n_in,
                              void* d_out, int out_size, void* d_ws, size_t ws_size,
                              hipStream_t stream) {
    const float* x        = (const float*)d_in[0];
    const float* wave_w   = (const float*)d_in[1];
    const float* wave_b   = (const float*)d_in[2];
    const float* wave_g   = (const float*)d_in[3];
    const float* kernel_w = (const float*)d_in[4];
    const float* kernel_b = (const float*)d_in[5];
    const float* kernel_g = (const float*)d_in[6];
    const float* out_w    = (const float*)d_in[7];
    float* out = (float*)d_out;

    auto align256 = [](size_t v) { return (v + 255) & ~(size_t)255; };
    char* ws = (char*)d_ws;
    size_t off = 0;
    bf16* xb16   = (bf16*)(ws + off); off = align256(off + (size_t)NT_ * C_ * sizeof(bf16));
    bf16* Wf     = (bf16*)(ws + off); off = align256(off + (size_t)NF_ * C_ * sizeof(bf16));
    bf16* Wo16   = (bf16*)(ws + off); off = align256(off + (size_t)C_ * C_ * sizeof(bf16));
    float* freq  = (float*)(ws + off); off = align256(off + (size_t)NT_ * H_ * sizeof(float));
    float* phase = (float*)(ws + off); off = align256(off + (size_t)NT_ * H_ * sizeof(float));
    float* kern  = (float*)(ws + off); off = align256(off + (size_t)NT_ * H_ * K_ * sizeof(float));
    bf16* hidden = (bf16*)(ws + off); off = align256(off + (size_t)NT_ * C_ * sizeof(bf16));
    (void)ws_size; (void)in_sizes; (void)n_in; (void)out_size;

    cvt_bf16<<<4096, 256, 0, stream>>>(x, xb16, NT_ * C_);
    cvt_bf16<<<128,  256, 0, stream>>>(wave_w,   Wf,            32 * C_);
    cvt_bf16<<<448,  256, 0, stream>>>(kernel_w, Wf + 32 * C_, 112 * C_);
    cvt_bf16<<<4096, 256, 0, stream>>>(out_w,    Wo16,         C_ * C_);

    gemm_front<<<NT_ / 64, 128, 0, stream>>>(xb16, Wf, wave_b, wave_g,
                                             kernel_b, kernel_g, freq, phase, kern);

    gather_conv<<<(NT_ * H_) / 8, 256, 0, stream>>>(x, freq, phase, kern, hidden);

    gemm_out<<<(NT_ / 128) * (C_ / 128), 256, 0, stream>>>(hidden, Wo16, out);
}
